// Attention_41712722379413
// MI455X (gfx1250) — compile-verified, matrix operand order checked by reading, not run on hardware
//
#include <hip/hip_runtime.h>
#include <hip/hip_bf16.h>
#include <stdint.h>

typedef __attribute__((ext_vector_type(16))) _Float16 v16h;
typedef __attribute__((ext_vector_type(8)))  _Float16 v8h;
typedef __attribute__((ext_vector_type(8)))  float    v8f;
typedef __attribute__((ext_vector_type(4)))  float    v4f;

#define BATCH  4
#define SEQ    2048
#define CDIM   768
#define NHEADS 12
#define DH     64
#define QKVN   (3 * CDIM)          // 2304
#define PERBUF ((size_t)BATCH * NHEADS * SEQ * DH)   // 6291456 halves per q/k/vt/o buffer

// ---------------------------------------------------------------------------
// WMMA helper: D = A(16x32 f16) * B(32x16 f16) + C(16x16 f32)
// ---------------------------------------------------------------------------
__device__ __forceinline__ v8f wmma_f16(v16h a, v16h b, v8f c) {
    return __builtin_amdgcn_wmma_f32_16x16x32_f16(
        /*neg_a=*/false, a, /*neg_b=*/false, b,
        /*c_mod=*/(short)0, c, /*reuse_a=*/false, /*reuse_b=*/false);
}

// A/B fragment loader from f16 row-major source (row stride ld, in elements).
// ISA 16-bit 16x32 layout: lane = row (M or N), half = lane>>4 selects K groups:
//   elems 0..7  <- K = 8*half .. +7 ; elems 8..15 <- K = 16+8*half .. +7
__device__ __forceinline__ v16h load_frag_f16(const _Float16* p0, int ld, int lane) {
    int row = lane & 15, half = lane >> 4;
    const _Float16* p = p0 + (size_t)row * ld + 8 * half;
    v8h a = *(const v8h*)p;
    v8h b = *(const v8h*)(p + 16);
    v16h r;
#pragma unroll
    for (int i = 0; i < 8; ++i) { r[i] = a[i]; r[8 + i] = b[i]; }
    return r;
}

// Same, but source is f32 (converted to f16 in registers).
__device__ __forceinline__ v16h load_frag_f32(const float* p0, int ld, int lane) {
    int row = lane & 15, half = lane >> 4;
    const float* p = p0 + (size_t)row * ld + 8 * half;
    v4f a0 = *(const v4f*)p;
    v4f a1 = *(const v4f*)(p + 4);
    v4f b0 = *(const v4f*)(p + 16);
    v4f b1 = *(const v4f*)(p + 20);
    v16h r;
#pragma unroll
    for (int i = 0; i < 4; ++i) {
        r[i]      = (_Float16)a0[i];
        r[4 + i]  = (_Float16)a1[i];
        r[8 + i]  = (_Float16)b0[i];
        r[12 + i] = (_Float16)b1[i];
    }
    return r;
}

// LDS byte offset of a shared-memory pointer (generic-pointer low 32 bits map
// directly to the LDS offset per the gfx1250 aperture rules).
__device__ __forceinline__ uint32_t lds_off(const void* p) {
    return (uint32_t)(uintptr_t)p;
}

// ---------------------------------------------------------------------------
// Kernel 1: qkv = x @ w_qkv^T, scattered to per-head q, k, and transposed v.
// One wave computes a 32(M) x 64(N) tile (B fragments reused 2x).
// ---------------------------------------------------------------------------
__global__ __launch_bounds__(128) void qkv_gemm_kernel(
    const float* __restrict__ x, const float* __restrict__ wqkv,
    _Float16* __restrict__ qh, _Float16* __restrict__ kh, _Float16* __restrict__ vt)
{
    const int lane = threadIdx.x & 31;
    const int wid  = (blockIdx.x * blockDim.x + threadIdx.x) >> 5;
    const int NT   = QKVN / 64;          // 36 n-tiles
    const int mt   = wid / NT;
    const int nt   = wid - mt * NT;
    const int m0   = mt * 32;
    const int n0   = nt * 64;

    v8f acc[2][4] = {};
    for (int k0 = 0; k0 < CDIM; k0 += 32) {
        __builtin_prefetch(x + (size_t)m0 * CDIM + k0 + 64, 0, 1);
        v16h a0 = load_frag_f32(x + (size_t)m0 * CDIM + k0,        CDIM, lane);
        v16h a1 = load_frag_f32(x + (size_t)(m0 + 16) * CDIM + k0, CDIM, lane);
#pragma unroll
        for (int t = 0; t < 4; ++t) {
            v16h b = load_frag_f32(wqkv + (size_t)(n0 + t * 16) * CDIM + k0, CDIM, lane);
            acc[0][t] = wmma_f16(a0, b, acc[0][t]);
            acc[1][t] = wmma_f16(a1, b, acc[1][t]);
        }
    }

    // Epilogue scatter. n-tile (64 cols) maps to exactly one of q/k/v and one head.
    const int b     = m0 / SEQ;
    const int nbase = m0 - b * SEQ;
    const int sel   = n0 / CDIM;                 // 0=q, 1=k, 2=v
    const int h     = (n0 - sel * CDIM) / DH;
    const int half  = lane >> 4, col = lane & 15;
    const size_t bh = (size_t)(b * NHEADS + h);
#pragma unroll
    for (int u = 0; u < 2; ++u) {
#pragma unroll
        for (int t = 0; t < 4; ++t) {
            const int d = t * 16 + col;
#pragma unroll
            for (int j = 0; j < 8; ++j) {
                const int npos = nbase + u * 16 + j + 8 * half;
                const _Float16 hv = (_Float16)acc[u][t][j];
                if (sel == 0)       qh[(bh * SEQ + npos) * DH + d] = hv;
                else if (sel == 1)  kh[(bh * SEQ + npos) * DH + d] = hv;
                else                vt[(bh * DH + d) * SEQ + npos] = hv;  // transposed
            }
        }
    }
}

// ---------------------------------------------------------------------------
// Kernel 2: RoPE applied in-place to q and k (pairs along head dim).
// ---------------------------------------------------------------------------
__global__ void rope_kernel(_Float16* __restrict__ qh, _Float16* __restrict__ kh,
                            const float* __restrict__ cosb, const float* __restrict__ sinb)
{
    const size_t PER = (size_t)BATCH * NHEADS * SEQ * (DH / 2);
    size_t idx = (size_t)blockIdx.x * blockDim.x + threadIdx.x;
    if (idx >= 2 * PER) return;
    _Float16* buf = (idx < PER) ? qh : kh;
    size_t p = (idx < PER) ? idx : idx - PER;
    const int i  = (int)(p & 31);                // freq index (Dh/2 = 32)
    const int n  = (int)((p >> 5) & (SEQ - 1));  // sequence position
    const size_t bhn = p >> 5;                   // (bh*SEQ + n)
    const size_t off = bhn * DH + 2 * i;
    const float a  = (float)buf[off];
    const float b2 = (float)buf[off + 1];
    const float c = cosb[n * (DH / 2) + i];
    const float s = sinb[n * (DH / 2) + i];
    buf[off]     = (_Float16)(a * c - b2 * s);
    buf[off + 1] = (_Float16)(a * s + b2 * c);
}

// ---------------------------------------------------------------------------
// Kernel 3: causal flash attention.
// Block = 64 query rows (4 waves x 16 rows), uniform key loop over blocks of
// 32 keys. K (32x64) and V^T (64x32) tiles are staged into LDS once per block
// via the CDNA5 async-to-LDS path (ASYNCcnt), then all 4 waves read fragments
// from LDS. Softmax stats via 16-lane shuffles; P re-laid-out through a
// per-wave LDS tile (DScnt) into an A fragment for the PV WMMAs.
// ---------------------------------------------------------------------------
__global__ __launch_bounds__(128) void flash_attn_kernel(
    const _Float16* __restrict__ qh, const _Float16* __restrict__ kh,
    const _Float16* __restrict__ vt, _Float16* __restrict__ o)
{
    __shared__ __align__(16) _Float16 kbuf[32 * DH];      // 32 keys x 64 d   (4 KB)
    __shared__ __align__(16) _Float16 vbuf[DH * 32];      // 64 d   x 32 keys (4 KB)
    __shared__ __align__(16) _Float16 pbuf[4][16 * 32];   // per-wave P tile  (4 KB)

    const int tid  = threadIdx.x;
    const int lane = tid & 31;
    const int w    = tid >> 5;
    const int qt   = blockIdx.x & 31;
    const int bh   = blockIdx.x >> 5;
    const int qb   = qt * 64;
    const int q0   = qb + w * 16;
    const int half = lane >> 4, col = lane & 15;

    const _Float16* Q = qh + (size_t)bh * SEQ * DH;
    const _Float16* K = kh + (size_t)bh * SEQ * DH;
    const _Float16* V = vt + (size_t)bh * DH * SEQ;

    const v16h aq0 = load_frag_f16(Q + (size_t)q0 * DH,      DH, lane);
    const v16h aq1 = load_frag_f16(Q + (size_t)q0 * DH + 32, DH, lane);

    v8f acc[4] = {};
    float mrun[8], lrun[8];
#pragma unroll
    for (int j = 0; j < 8; ++j) { mrun[j] = -1e30f; lrun[j] = 0.0f; }

    const int kend = qb + 64;               // block-uniform causal horizon
    for (int kb = 0; kb < kend; kb += 32) {
        __syncthreads();                    // prior iteration's LDS reads done

        // ---- async-stage K tile (32x64) and V^T tile (64x32) into LDS ----
#pragma unroll
        for (int r = 0; r < 2; ++r) {
            const int c = tid + r * 128;    // 256 16-byte chunks per tile
            const int krow = c >> 3, koff = (c & 7) * 8;
            uint64_t gk = (uint64_t)(uintptr_t)(K + (size_t)(kb + krow) * DH + koff);
            uint32_t lk = lds_off(kbuf + krow * DH + koff);
            asm volatile("global_load_async_to_lds_b128 %0, %1, off"
                         :: "v"(lk), "v"(gk) : "memory");
            const int vrow = c >> 2, voff = (c & 3) * 8;
            uint64_t gv = (uint64_t)(uintptr_t)(V + (size_t)vrow * SEQ + kb + voff);
            uint32_t lv = lds_off(vbuf + vrow * 32 + voff);
            asm volatile("global_load_async_to_lds_b128 %0, %1, off"
                         :: "v"(lv), "v"(gv) : "memory");
        }
        asm volatile("s_wait_asynccnt 0" ::: "memory");
        __syncthreads();                    // tiles visible to all waves

        // ---- S = Q K^T for 32 keys (two 16-key tiles), K from LDS ----
        v16h bk00 = load_frag_f16(kbuf,                DH, lane);
        v16h bk01 = load_frag_f16(kbuf + 32,           DH, lane);
        v16h bk10 = load_frag_f16(kbuf + 16 * DH,      DH, lane);
        v16h bk11 = load_frag_f16(kbuf + 16 * DH + 32, DH, lane);
        v8f s0 = {}; s0 = wmma_f16(aq0, bk00, s0); s0 = wmma_f16(aq1, bk01, s0);
        v8f s1 = {}; s1 = wmma_f16(aq0, bk10, s1); s1 = wmma_f16(aq1, bk11, s1);

        // ---- scale + causal mask + online softmax ----
        float p0[8], p1[8];
#pragma unroll
        for (int j = 0; j < 8; ++j) {
            const int row = q0 + j + 8 * half;
            float v0 = s0[j] * 0.125f + ((kb + col)      <= row ? 0.0f : -1e9f);
            float v1 = s1[j] * 0.125f + ((kb + 16 + col) <= row ? 0.0f : -1e9f);
            float mx = fmaxf(v0, v1);
#pragma unroll
            for (int off = 1; off < 16; off <<= 1)
                mx = fmaxf(mx, __shfl_xor(mx, off, 32));
            const float mnew = fmaxf(mrun[j], mx);
            const float e0 = __expf(v0 - mnew);
            const float e1 = __expf(v1 - mnew);
            float ls = e0 + e1;
#pragma unroll
            for (int off = 1; off < 16; off <<= 1)
                ls += __shfl_xor(ls, off, 32);
            const float alpha = __expf(mrun[j] - mnew);
            lrun[j] = lrun[j] * alpha + ls;
            mrun[j] = mnew;
            p0[j] = e0; p1[j] = e1;
#pragma unroll
            for (int t = 0; t < 4; ++t) acc[t][j] *= alpha;
        }

        // ---- re-layout P (C layout -> A fragment) through per-wave LDS ----
        _Float16* pp = pbuf[w];
#pragma unroll
        for (int j = 0; j < 8; ++j) {
            const int r = j + 8 * half;
            pp[r * 32 + col]      = (_Float16)p0[j];
            pp[r * 32 + 16 + col] = (_Float16)p1[j];
        }
        asm volatile("s_wait_dscnt 0" ::: "memory");
        const v16h ap = load_frag_f16(pp, 32, lane);

        // ---- O += P * V^T, V^T from LDS (4 output d-tiles of 16) ----
#pragma unroll
        for (int t = 0; t < 4; ++t) {
            v16h bv = load_frag_f16(vbuf + (size_t)(t * 16) * 32, 32, lane);
            acc[t] = wmma_f16(ap, bv, acc[t]);
        }
    }

    // ---- normalize and store O in [B, N, C] f16 layout for proj GEMM ----
    const int b = bh / NHEADS;
    const int h = bh - b * NHEADS;
#pragma unroll
    for (int j = 0; j < 8; ++j) {
        const float rinv = 1.0f / lrun[j];
        const size_t rowg = (size_t)b * SEQ + (q0 + j + 8 * half);
#pragma unroll
        for (int t = 0; t < 4; ++t)
            o[rowg * CDIM + h * DH + t * 16 + col] = (_Float16)(acc[t][j] * rinv);
    }
}

// ---------------------------------------------------------------------------
// Kernel 4: out = O @ w_proj^T + b_proj  (32x64 per-wave tile, f32 out)
// ---------------------------------------------------------------------------
__global__ __launch_bounds__(128) void proj_gemm_kernel(
    const _Float16* __restrict__ o, const float* __restrict__ wproj,
    const float* __restrict__ bias, float* __restrict__ out)
{
    const int lane = threadIdx.x & 31;
    const int wid  = (blockIdx.x * blockDim.x + threadIdx.x) >> 5;
    const int NT   = CDIM / 64;          // 12
    const int mt   = wid / NT;
    const int nt   = wid - mt * NT;
    const int m0   = mt * 32;
    const int n0   = nt * 64;

    v8f acc[2][4] = {};
    for (int k0 = 0; k0 < CDIM; k0 += 32) {
        v16h a0 = load_frag_f16(o + (size_t)m0 * CDIM + k0,        CDIM, lane);
        v16h a1 = load_frag_f16(o + (size_t)(m0 + 16) * CDIM + k0, CDIM, lane);
#pragma unroll
        for (int t = 0; t < 4; ++t) {
            v16h b = load_frag_f32(wproj + (size_t)(n0 + t * 16) * CDIM + k0, CDIM, lane);
            acc[0][t] = wmma_f16(a0, b, acc[0][t]);
            acc[1][t] = wmma_f16(a1, b, acc[1][t]);
        }
    }

    const int half = lane >> 4, col = lane & 15;
#pragma unroll
    for (int u = 0; u < 2; ++u) {
#pragma unroll
        for (int t = 0; t < 4; ++t) {
            const float bval = bias[n0 + t * 16 + col];
#pragma unroll
            for (int j = 0; j < 8; ++j)
                out[(size_t)(m0 + u * 16 + j + 8 * half) * CDIM + n0 + t * 16 + col] =
                    acc[u][t][j] + bval;
        }
    }
}

// ---------------------------------------------------------------------------
// Launch: inputs are {x, freqs_cos, freqs_sin, mask, w_qkv, w_proj, b_proj}.
// Workspace: q | k | v^T | O, each PERBUF f16 elements (~50 MB total).
// ---------------------------------------------------------------------------
extern "C" void kernel_launch(void* const* d_in, const int* in_sizes, int n_in,
                              void* d_out, int out_size, void* d_ws, size_t ws_size,
                              hipStream_t stream) {
    (void)in_sizes; (void)n_in; (void)out_size; (void)ws_size;
    const float* x     = (const float*)d_in[0];
    const float* fcos  = (const float*)d_in[1];
    const float* fsin  = (const float*)d_in[2];
    const float* wqkv  = (const float*)d_in[4];
    const float* wproj = (const float*)d_in[5];
    const float* bproj = (const float*)d_in[6];
    float* out = (float*)d_out;

    _Float16* ws = (_Float16*)d_ws;
    _Float16* qh = ws;
    _Float16* kh = ws + PERBUF;
    _Float16* vt = ws + 2 * PERBUF;
    _Float16* oh = ws + 3 * PERBUF;

    // 256 m-tiles (of 32) * 36 n-tiles = 9216 waves / 4 per block
    qkv_gemm_kernel<<<2304, 128, 0, stream>>>(x, wqkv, qh, kh, vt);
    // 2 * 3,145,728 rotation pairs
    rope_kernel<<<24576, 256, 0, stream>>>(qh, kh, fcos, fsin);
    // B*H * 32 q-tiles of 64 rows
    flash_attn_kernel<<<1536, 128, 0, stream>>>(qh, kh, vt, oh);
    // 256 m-tiles * 12 n-tiles = 3072 waves / 4 per block
    proj_gemm_kernel<<<768, 128, 0, stream>>>(oh, wproj, bproj, out);
}
